// MultiHeadSelfAttention_41772851921165
// MI455X (gfx1250) — compile-verified, matrix operand order checked by reading, not run on hardware
//
#include <hip/hip_runtime.h>

// ---------------------------------------------------------------------------
// MHSA for MI455X (gfx1250): fp16 WMMA everywhere, f32 accumulate.
//   B=2, T=4096, C=768, H=12, Dh=64
// Round 3: LDS-only workgroup barrier (s_wait_dscnt + split barrier) so the
// double-buffer global prefetch is NOT flushed at the barrier (previously
// __syncthreads emitted s_wait_loadcnt_dscnt 0x0, serializing the prefetch).
// ---------------------------------------------------------------------------

typedef _Float16 h16;
typedef __attribute__((ext_vector_type(16))) _Float16 v16h;
typedef __attribute__((ext_vector_type(8)))  _Float16 v8h;
typedef __attribute__((ext_vector_type(4)))  _Float16 v4h;
typedef __attribute__((ext_vector_type(8)))  float    v8f;
typedef __attribute__((ext_vector_type(4)))  float    v4f;

#define WMMA_F16(a, b, c) \
  __builtin_amdgcn_wmma_f32_16x16x32_f16(false, (a), false, (b), (short)0, (c), false, false)

#define TSEQ   4096
#define DMODEL 768
#define NHEAD  12
#define DHEAD  64
#define MROWS  8192     // B*T

// Workgroup barrier that only fences LDS: wait for this wave's outstanding DS
// ops, then split-barrier. Crucially does NOT wait LOADcnt, so in-flight
// global prefetch loads keep flying across the barrier into the WMMA phase.
__device__ __forceinline__ void wg_barrier_lds() {
  asm volatile(
      "s_wait_dscnt 0x0\n\t"
      "s_barrier_signal -1\n\t"
      "s_barrier_wait -1" ::: "memory");
}

// --- WMMA fragment loader (A or B operand) from a K-contiguous row-major
// buffer. Per ISA 7.12.2 (16-bit A 16x32): lane<16 -> row=lane, K {0..7,16..23};
// lane>=16 -> row=lane-16, K {8..15,24..31}. Two 16-byte loads.
__device__ __forceinline__ v16h load_frag(const h16* base, int stride_h) {
  const int lane = threadIdx.x & 31;
  const h16* p = base + (lane & 15) * stride_h + ((lane >> 4) << 3);
  v8h lo = *(const v8h*)(p);
  v8h hi = *(const v8h*)(p + 16);
  v16h f;
#pragma unroll
  for (int i = 0; i < 8; ++i) { f[i] = lo[i]; f[i + 8] = hi[i]; }
  return f;
}

// --- fp32 -> fp16 conversion, 4 elements/thread -----------------------------
__global__ __launch_bounds__(256) void cvt_f32_f16_kernel(const float* __restrict__ s,
                                                          h16* __restrict__ d, int n4) {
  int i = blockIdx.x * 256 + threadIdx.x;
  if (i < n4) {
    v4f v = *(const v4f*)(s + (size_t)i * 4);
    v4h o = {(h16)v[0], (h16)v[1], (h16)v[2], (h16)v[3]};
    *(v4h*)(d + (size_t)i * 4) = o;
  }
}

// ---------------------------------------------------------------------------
// Shared GEMM mainloop pieces (128x128 block, 8 waves x (64x32) tiles,
// K-step 32, double-buffered staging).  LDS stride 40 halves = 80 B (16B
// aligned ds_load_b128, conflict-free across the 16 rows of a fragment).
// ---------------------------------------------------------------------------
__device__ __forceinline__ void gemm_fetch(const h16* g, int row0, int seg,
                                           v8h r[2]) {
  r[0] = *(const v8h*)(g + (size_t)row0 * DMODEL + seg);
  r[1] = *(const v8h*)(g + (size_t)(row0 + 64) * DMODEL + seg);
}
__device__ __forceinline__ void gemm_stage(h16* s, int row0, int seg,
                                           const v8h r[2]) {
  *(v8h*)(s + row0 * 40 + seg) = r[0];
  *(v8h*)(s + (row0 + 64) * 40 + seg) = r[1];
}

// ---------------------------------------------------------------------------
// GEMM1: QKV = x[8192x768] * w_qkv^T[768x2304] + b ; scatter into Q/K/V
// [B,H,T,Dh] fp16, Q pre-scaled by 1/sqrt(Dh). Grid (64, 18), 256 threads.
// ---------------------------------------------------------------------------
__global__ __launch_bounds__(256) void qkv_gemm_kernel(
    const h16* __restrict__ X, const h16* __restrict__ W,
    const float* __restrict__ bias, h16* __restrict__ Qo,
    h16* __restrict__ Ko, h16* __restrict__ Vo) {
  __shared__ h16 sA[128 * 40];
  __shared__ h16 sB[128 * 40];

  const int m0 = blockIdx.x * 128;
  const int n0 = blockIdx.y * 128;
  const int wave = threadIdx.x >> 5;
  const int lane = threadIdx.x & 31;
  const int wm = (wave >> 2) * 64;  // 0 / 64
  const int wn = (wave & 3) * 32;   // 0..96
  const int t = threadIdx.x;
  const int row0 = t >> 2;          // 0..63 (chunk pair: row0, row0+64)
  const int seg  = (t & 3) * 8;

  const h16* Ag = X + (size_t)m0 * DMODEL;
  const h16* Bg = W + (size_t)n0 * DMODEL;

  v8f acc[4][2] = {};
  v8h ra[2], rb[2], ra2[2], rb2[2];
  gemm_fetch(Ag, row0, seg, ra);
  gemm_fetch(Bg, row0, seg, rb);

  for (int kt = 0; kt < DMODEL; kt += 32) {
    wg_barrier_lds();
    gemm_stage(sA, row0, seg, ra);
    gemm_stage(sB, row0, seg, rb);
    const int ktn = (kt + 32 < DMODEL) ? kt + 32 : kt;  // harmless refetch tail
    gemm_fetch(Ag + ktn, row0, seg, ra2);
    gemm_fetch(Bg + ktn, row0, seg, rb2);
    wg_barrier_lds();

    v16h aF[4], bF[2];
#pragma unroll
    for (int i = 0; i < 4; ++i) aF[i] = load_frag(sA + (wm + i * 16) * 40, 40);
#pragma unroll
    for (int j = 0; j < 2; ++j) bF[j] = load_frag(sB + (wn + j * 16) * 40, 40);
#pragma unroll
    for (int i = 0; i < 4; ++i)
#pragma unroll
      for (int j = 0; j < 2; ++j) acc[i][j] = WMMA_F16(aF[i], bF[j], acc[i][j]);

#pragma unroll
    for (int u = 0; u < 2; ++u) { ra[u] = ra2[u]; rb[u] = rb2[u]; }
  }

  const int col = lane & 15;
  const int rb8 = (lane >> 4) * 8;
#pragma unroll
  for (int i = 0; i < 4; ++i) {
#pragma unroll
    for (int j = 0; j < 2; ++j) {
      int n = n0 + wn + j * 16 + col;       // 0..2303
      float bs = bias[n];
      int which = n / DMODEL;
      int rem = n - which * DMODEL;
      int hh = rem >> 6, dd = rem & 63;
      h16* dst = (which == 0) ? Qo : (which == 1) ? Ko : Vo;
      float sc = (which == 0) ? 0.125f : 1.0f;  // 1/sqrt(64) pre-applied to Q
#pragma unroll
      for (int r = 0; r < 8; ++r) {
        int m = m0 + wm + i * 16 + rb8 + r;
        int bb = m >> 12, tt = m & (TSEQ - 1);
        size_t off = ((((size_t)bb * NHEAD + hh) * TSEQ) + tt) * DHEAD + dd;
        dst[off] = (h16)((acc[i][j][r] + bs) * sc);
      }
    }
  }
}

// ---------------------------------------------------------------------------
// Flash attention: grid (T/128, B*H). 8 waves x 16 query rows, 64-key blocks.
// Online softmax entirely in the WMMA C/D register layout: component r of an
// accumulator maps to row rb8+r, so running (m, l, rescale) stats live in
// registers; row reductions are 4 shfl_xor steps inside each 16-lane
// half-group (the set of lanes sharing those 8 rows).
// ---------------------------------------------------------------------------
__global__ __launch_bounds__(256) void flash_attn_kernel(
    const h16* __restrict__ Q, const h16* __restrict__ K,
    const h16* __restrict__ V, const int* __restrict__ mask,
    h16* __restrict__ Aout) {
  __shared__ h16   sK[64 * 72];       // [key][d]
  __shared__ h16   sVt[64 * 72];      // [d][key]  (V transposed)
  __shared__ h16   sP[8][16 * 72];    // per-wave exp(S) in A-frag layout
  __shared__ float sMadd[64];

  const int qb = blockIdx.x;
  const int bh = blockIdx.y;
  const int b  = bh / NHEAD;
  const int h  = bh - b * NHEAD;
  const h16* Qb = Q + (size_t)bh * TSEQ * DHEAD;
  const h16* Kb = K + (size_t)bh * TSEQ * DHEAD;
  const h16* Vb = V + (size_t)bh * TSEQ * DHEAD;

  const int wave = threadIdx.x >> 5;
  const int lane = threadIdx.x & 31;
  const int col  = lane & 15;
  const int rb8  = (lane >> 4) * 8;
  const int q0   = qb * 128 + wave * 16;
  const int t    = threadIdx.x;
  const int row0 = t >> 3;            // 0..31 (chunk pair: row0, row0+32)
  const int seg  = (t & 7) * 8;

  // preload Q fragments (global, K-contiguous rows)
  v16h qF[2];
  {
    const h16* qbase = Qb + (size_t)q0 * DHEAD;
    qF[0] = load_frag(qbase, DHEAD);       // d 0..31
    qF[1] = load_frag(qbase + 32, DHEAD);  // d 32..63
  }

  v8f o[4] = {};
  float m8[8], l8[8];
#pragma unroll
  for (int r = 0; r < 8; ++r) { m8[r] = -1e30f; l8[r] = 0.0f; }

  // double-buffered K/V staging registers
  v8h rk[2], rv[2], rk2[2], rv2[2];
  int rm, rm2;
  {
    rk[0] = *(const v8h*)(Kb + (size_t)row0 * DHEAD + seg);
    rk[1] = *(const v8h*)(Kb + (size_t)(row0 + 32) * DHEAD + seg);
    rv[0] = *(const v8h*)(Vb + (size_t)row0 * DHEAD + seg);
    rv[1] = *(const v8h*)(Vb + (size_t)(row0 + 32) * DHEAD + seg);
    rm = (t < 64) ? mask[b * TSEQ + t] : 1;
  }

  for (int k0 = 0; k0 < TSEQ; k0 += 64) {
    wg_barrier_lds();
    // stage current tile from registers
    *(v8h*)(sK + row0 * 72 + seg)        = rk[0];
    *(v8h*)(sK + (row0 + 32) * 72 + seg) = rk[1];
#pragma unroll
    for (int e = 0; e < 8; ++e) sVt[(seg + e) * 72 + row0]      = rv[0][e];
#pragma unroll
    for (int e = 0; e < 8; ++e) sVt[(seg + e) * 72 + row0 + 32] = rv[1][e];
    if (t < 64) sMadd[t] = (rm == 0) ? -1e9f : 0.0f;

    // prefetch next tile (overlaps with WMMA phase below)
    {
      const int kn = (k0 + 64 < TSEQ) ? k0 + 64 : k0;
      rk2[0] = *(const v8h*)(Kb + (size_t)(kn + row0) * DHEAD + seg);
      rk2[1] = *(const v8h*)(Kb + (size_t)(kn + row0 + 32) * DHEAD + seg);
      rv2[0] = *(const v8h*)(Vb + (size_t)(kn + row0) * DHEAD + seg);
      rv2[1] = *(const v8h*)(Vb + (size_t)(kn + row0 + 32) * DHEAD + seg);
      rm2 = (t < 64) ? mask[b * TSEQ + kn + t] : 1;
    }
    wg_barrier_lds();

    // S = Q * K^T (+ mask addend), kept in C/D register layout
    v8f s[4];
#pragma unroll
    for (int nt = 0; nt < 4; ++nt) {
      v8f z = {};
      z = WMMA_F16(qF[0], load_frag(sK + nt * 16 * 72, 72), z);
      z = WMMA_F16(qF[1], load_frag(sK + nt * 16 * 72 + 32, 72), z);
      float madd = sMadd[nt * 16 + col];
#pragma unroll
      for (int r = 0; r < 8; ++r) z[r] += madd;
      s[nt] = z;
    }

    // --- online softmax, register resident ---
    float mloc[8];
#pragma unroll
    for (int r = 0; r < 8; ++r)
      mloc[r] = fmaxf(fmaxf(s[0][r], s[1][r]), fmaxf(s[2][r], s[3][r]));
#pragma unroll
    for (int xm = 1; xm <= 8; xm <<= 1)
#pragma unroll
      for (int r = 0; r < 8; ++r)
        mloc[r] = fmaxf(mloc[r], __shfl_xor(mloc[r], xm, 32));

    float fac[8];
#pragma unroll
    for (int r = 0; r < 8; ++r) {
      float mn = fmaxf(m8[r], mloc[r]);
      fac[r] = __expf(m8[r] - mn);
      m8[r] = mn;
    }
    float lloc[8];
#pragma unroll
    for (int r = 0; r < 8; ++r) lloc[r] = 0.0f;
#pragma unroll
    for (int nt = 0; nt < 4; ++nt)
#pragma unroll
      for (int r = 0; r < 8; ++r) {
        float p = __expf(s[nt][r] - m8[r]);
        s[nt][r] = p;
        lloc[r] += p;
      }
#pragma unroll
    for (int xm = 1; xm <= 8; xm <<= 1)
#pragma unroll
      for (int r = 0; r < 8; ++r) lloc[r] += __shfl_xor(lloc[r], xm, 32);
#pragma unroll
    for (int r = 0; r < 8; ++r) l8[r] = l8[r] * fac[r] + lloc[r];

    // write P (fp16) into per-wave A-fragment staging; same-wave LDS is
    // in-order so no barrier needed before the fragment loads below.
    h16* pw = sP[wave];
#pragma unroll
    for (int nt = 0; nt < 4; ++nt)
#pragma unroll
      for (int r = 0; r < 8; ++r)
        pw[(rb8 + r) * 72 + nt * 16 + col] = (h16)s[nt][r];

    // rescale running O, then O += P * V
#pragma unroll
    for (int dt = 0; dt < 4; ++dt)
#pragma unroll
      for (int r = 0; r < 8; ++r) o[dt][r] *= fac[r];

    v16h a0 = load_frag(pw, 72);        // keys 0..31
    v16h a1 = load_frag(pw + 32, 72);   // keys 32..63
#pragma unroll
    for (int dt = 0; dt < 4; ++dt) {
      o[dt] = WMMA_F16(a0, load_frag(sVt + dt * 16 * 72, 72), o[dt]);
      o[dt] = WMMA_F16(a1, load_frag(sVt + dt * 16 * 72 + 32, 72), o[dt]);
    }

#pragma unroll
    for (int u = 0; u < 2; ++u) { rk[u] = rk2[u]; rv[u] = rv2[u]; }
    rm = rm2;
  }

  // finalize: divide by l, store fp16 into [B,T,C] at column h*64
  float li[8];
#pragma unroll
  for (int r = 0; r < 8; ++r) li[r] = 1.0f / l8[r];
#pragma unroll
  for (int dt = 0; dt < 4; ++dt) {
#pragma unroll
    for (int r = 0; r < 8; ++r) {
      size_t m = (size_t)b * TSEQ + q0 + rb8 + r;
      Aout[m * DMODEL + h * DHEAD + dt * 16 + col] = (h16)(o[dt][r] * li[r]);
    }
  }
}

// ---------------------------------------------------------------------------
// GEMM2: out = attn[8192x768] * w_out^T[768x768] + b_out  (fp32 output)
// Grid (64, 6), 256 threads.
// ---------------------------------------------------------------------------
__global__ __launch_bounds__(256) void out_gemm_kernel(
    const h16* __restrict__ A, const h16* __restrict__ W,
    const float* __restrict__ bias, float* __restrict__ out) {
  __shared__ h16 sA[128 * 40];
  __shared__ h16 sB[128 * 40];

  const int m0 = blockIdx.x * 128;
  const int n0 = blockIdx.y * 128;
  const int wave = threadIdx.x >> 5;
  const int lane = threadIdx.x & 31;
  const int wm = (wave >> 2) * 64;
  const int wn = (wave & 3) * 32;
  const int t = threadIdx.x;
  const int row0 = t >> 2;
  const int seg  = (t & 3) * 8;

  const h16* Ag = A + (size_t)m0 * DMODEL;
  const h16* Bg = W + (size_t)n0 * DMODEL;

  v8f acc[4][2] = {};
  v8h ra[2], rb[2], ra2[2], rb2[2];
  gemm_fetch(Ag, row0, seg, ra);
  gemm_fetch(Bg, row0, seg, rb);

  for (int kt = 0; kt < DMODEL; kt += 32) {
    wg_barrier_lds();
    gemm_stage(sA, row0, seg, ra);
    gemm_stage(sB, row0, seg, rb);
    const int ktn = (kt + 32 < DMODEL) ? kt + 32 : kt;
    gemm_fetch(Ag + ktn, row0, seg, ra2);
    gemm_fetch(Bg + ktn, row0, seg, rb2);
    wg_barrier_lds();

    v16h aF[4], bF[2];
#pragma unroll
    for (int i = 0; i < 4; ++i) aF[i] = load_frag(sA + (wm + i * 16) * 40, 40);
#pragma unroll
    for (int j = 0; j < 2; ++j) bF[j] = load_frag(sB + (wn + j * 16) * 40, 40);
#pragma unroll
    for (int i = 0; i < 4; ++i)
#pragma unroll
      for (int j = 0; j < 2; ++j) acc[i][j] = WMMA_F16(aF[i], bF[j], acc[i][j]);

#pragma unroll
    for (int u = 0; u < 2; ++u) { ra[u] = ra2[u]; rb[u] = rb2[u]; }
  }

  const int col = lane & 15;
  const int rb8 = (lane >> 4) * 8;
#pragma unroll
  for (int i = 0; i < 4; ++i) {
#pragma unroll
    for (int j = 0; j < 2; ++j) {
      int n = n0 + wn + j * 16 + col;
      float bs = bias[n];
#pragma unroll
      for (int r = 0; r < 8; ++r) {
        int m = m0 + wm + i * 16 + rb8 + r;
        out[(size_t)m * DMODEL + n] = acc[i][j][r] + bs;
      }
    }
  }
}

// ---------------------------------------------------------------------------
extern "C" void kernel_launch(void* const* d_in, const int* in_sizes, int n_in,
                              void* d_out, int out_size, void* d_ws, size_t ws_size,
                              hipStream_t stream) {
  const float* x     = (const float*)d_in[0];
  const float* w_qkv = (const float*)d_in[1];
  const float* b_qkv = (const float*)d_in[2];
  const float* w_out = (const float*)d_in[3];
  const float* b_out = (const float*)d_in[4];
  const int*   mask  = (const int*)d_in[5];
  float* out = (float*)d_out;

  const size_t N_X    = (size_t)MROWS * DMODEL;       // 6,291,456
  const size_t N_WQKV = (size_t)3 * DMODEL * DMODEL;  // 1,769,472
  const size_t N_WOUT = (size_t)DMODEL * DMODEL;      //   589,824

  h16* xf = (h16*)d_ws;
  h16* wq = xf + N_X;
  h16* wo = wq + N_WQKV;
  h16* Qb = wo + N_WOUT;
  h16* Kb = Qb + N_X;
  h16* Vb = Kb + N_X;
  h16* Ab = Vb + N_X;

  cvt_f32_f16_kernel<<<(int)(N_X / 4 / 256),    256, 0, stream>>>(x, xf, (int)(N_X / 4));
  cvt_f32_f16_kernel<<<(int)(N_WQKV / 4 / 256), 256, 0, stream>>>(w_qkv, wq, (int)(N_WQKV / 4));
  cvt_f32_f16_kernel<<<(int)(N_WOUT / 4 / 256), 256, 0, stream>>>(w_out, wo, (int)(N_WOUT / 4));

  qkv_gemm_kernel<<<dim3(MROWS / 128, (3 * DMODEL) / 128), 256, 0, stream>>>(
      xf, wq, b_qkv, Qb, Kb, Vb);

  flash_attn_kernel<<<dim3(TSEQ / 128, 2 * NHEAD), 256, 0, stream>>>(
      Qb, Kb, Vb, mask, Ab);

  out_gemm_kernel<<<dim3(MROWS / 128, DMODEL / 128), 256, 0, stream>>>(
      Ab, wo, b_out, out);
}
